// RelPartialLearnableMultiHeadAttn_8083128451277
// MI455X (gfx1250) — matrix operand order repaired and checked by reading, not compiled, not run on hardware
//
#include <hip/hip_runtime.h>

// ---------------------------------------------------------------------------
// Types for CDNA5 WMMA (wave32, v_wmma_f32_16x16x32_bf16)
// ---------------------------------------------------------------------------
typedef __bf16 bf16_t;
typedef __attribute__((ext_vector_type(16))) __bf16        v16bf;
typedef __attribute__((ext_vector_type(8)))  float         v8f;
typedef __attribute__((ext_vector_type(4)))  unsigned int  u32x4;
typedef __attribute__((ext_vector_type(4)))  float         f32x4;
typedef __attribute__((ext_vector_type(4)))  __bf16        v4bf;

static constexpr int Bb  = 4;
static constexpr int Qq  = 1024;
static constexpr int Dd  = 1024;
static constexpr int Hh  = 16;
static constexpr int DHh = 64;
static constexpr int RPAD = Qq + 16;      // r_head_k rows incl. zero pad (rel_shift)
static constexpr float SCALE = 0.125f;    // 1/sqrt(64)
static constexpr float NEG_INF = -1e30f;

union V16 { v16bf v; u32x4 q[2]; };

// A-fragment (16x32 bf16): lane M = lane&15, K = 16h + e (contiguous)
__device__ inline v16bf load_a_frag(const bf16_t* base, int lh) {
    V16 u;
    u.q[0] = *(const u32x4*)(base + 16 * lh);
    u.q[1] = *(const u32x4*)(base + 16 * lh + 8);
    return u.v;
}
// B-fragment (32x16 bf16): lane N = lane&15, K = 16*(lane>>4) + e  -> 32 contiguous bytes
// (caller pre-adds the 16*(lane>>4) element offset)
__device__ inline v16bf load_b_frag(const bf16_t* p) {
    V16 u;
    u.q[0] = *(const u32x4*)(p);
    u.q[1] = *(const u32x4*)(p + 8);
    return u.v;
}
__device__ inline v8f wmma_bf16(v16bf a, v16bf b, v8f c) {
    return __builtin_amdgcn_wmma_f32_16x16x32_bf16(
        /*neg_a=*/false, a, /*neg_b=*/false, b,
        /*c_mod=*/(short)0, c, /*reuse_a=*/false, /*reuse_b=*/false);
}
__device__ inline v8f v8zero() { v8f z = {0.f,0.f,0.f,0.f,0.f,0.f,0.f,0.f}; return z; }

// ---------------------------------------------------------------------------
// fp32 -> bf16 convert (vectorized x4)
// ---------------------------------------------------------------------------
__global__ __launch_bounds__(256) void cvt_k(const float* __restrict__ in,
                                             bf16_t* __restrict__ out, int n4) {
    int i = blockIdx.x * 256 + threadIdx.x;
    if (i < n4) {
        f32x4 f = ((const f32x4*)in)[i];
        v4bf o;
        o[0] = (bf16_t)f[0]; o[1] = (bf16_t)f[1];
        o[2] = (bf16_t)f[2]; o[3] = (bf16_t)f[3];
        ((v4bf*)out)[i] = o;
    }
}

__global__ __launch_bounds__(256) void zero_pad_k(bf16_t* __restrict__ rhk) {
    int idx = blockIdx.x * 256 + threadIdx.x;      // 16 heads * 16 pad rows * 64
    if (idx < Hh * 16 * DHh) {
        int h   = idx >> 10;                       // / (16*64)
        int rem = idx & 1023;
        rhk[((size_t)h * RPAD + Qq) * DHh + rem] = (bf16_t)0.0f;
    }
}

// ---------------------------------------------------------------------------
// GEMM: out[R x 1024] = A[R x 1024] @ W.T   (W stored torch-style [N x K])
// Block: 256 thr (8 waves), tile 128x128; wave tile 32(M) x 64(N); K step 32.
// MODE epilogues:
//   0: f32 row-major          1: Q-heads +two biases (bf16 x2, [B,H,Q,Dh])
//   2: K-heads bf16           3: V-heads transposed bf16 [B,H,Dh,Q]
//   4: r_head_k bf16 [H, Q+16, Dh]
// ---------------------------------------------------------------------------
template <int MODE>
__global__ __launch_bounds__(256) void gemm_k(
    const bf16_t* __restrict__ A, const bf16_t* __restrict__ W,
    float* __restrict__ outF, bf16_t* __restrict__ out0, bf16_t* __restrict__ out1,
    const float* __restrict__ bias0, const float* __restrict__ bias1) {

    const int lane = threadIdx.x & 31;
    const int wid  = threadIdx.x >> 5;
    const int wm = wid & 3, wn = wid >> 2;
    const int m0 = blockIdx.x * 128 + wm * 32;
    const int n0 = blockIdx.y * 128 + wn * 64;
    const int lrow = lane & 15, lh = lane >> 4;

    v8f acc[2][4];
#pragma unroll
    for (int mt = 0; mt < 2; ++mt)
#pragma unroll
        for (int nt = 0; nt < 4; ++nt) acc[mt][nt] = v8zero();

    const bf16_t* arow0 = A + (size_t)(m0 + lrow) * Dd;
    const bf16_t* arow1 = A + (size_t)(m0 + 16 + lrow) * Dd;
    const bf16_t* wrow[4];
#pragma unroll
    for (int nt = 0; nt < 4; ++nt)
        wrow[nt] = W + (size_t)(n0 + nt * 16 + lrow) * Dd + 16 * lh;

    for (int k0 = 0; k0 < Dd; k0 += 32) {
        if (k0 + 256 < Dd) {
            __builtin_prefetch(arow0 + k0 + 256, 0, 0);
            __builtin_prefetch(wrow[0] + k0 + 256, 0, 0);
        }
        v16bf a0 = load_a_frag(arow0 + k0, lh);
        v16bf a1 = load_a_frag(arow1 + k0, lh);
#pragma unroll
        for (int nt = 0; nt < 4; ++nt) {
            v16bf b = load_b_frag(wrow[nt] + k0);
            acc[0][nt] = wmma_bf16(a0, b, acc[0][nt]);
            acc[1][nt] = wmma_bf16(a1, b, acc[1][nt]);
        }
    }

#pragma unroll
    for (int mt = 0; mt < 2; ++mt) {
#pragma unroll
        for (int nt = 0; nt < 4; ++nt) {
#pragma unroll
            for (int i = 0; i < 8; ++i) {
                const int gr = m0 + mt * 16 + 8 * lh + i;   // D layout: row = vgpr + 8h
                const int gc = n0 + nt * 16 + lrow;         //           col = lane&15
                const float v = acc[mt][nt][i];
                if constexpr (MODE == 0) {
                    outF[(size_t)gr * Dd + gc] = v;
                } else if constexpr (MODE == 1) {
                    const int b = gr >> 10, q = gr & (Qq - 1);
                    const int h = gc >> 6, dh = gc & (DHh - 1);
                    const size_t idx = (((size_t)b * Hh + h) * Qq + q) * DHh + dh;
                    out0[idx] = (bf16_t)(v + bias0[gc]);
                    out1[idx] = (bf16_t)(v + bias1[gc]);
                } else if constexpr (MODE == 2) {
                    const int b = gr >> 10, q = gr & (Qq - 1);
                    const int h = gc >> 6, dh = gc & (DHh - 1);
                    out0[(((size_t)b * Hh + h) * Qq + q) * DHh + dh] = (bf16_t)v;
                } else if constexpr (MODE == 3) {           // V transposed [B,H,Dh,Q]
                    const int b = gr >> 10, q = gr & (Qq - 1);
                    const int h = gc >> 6, dh = gc & (DHh - 1);
                    out0[(((size_t)b * Hh + h) * DHh + dh) * Qq + q] = (bf16_t)v;
                } else {                                    // r_head_k [H, Q+16, Dh]
                    const int h = gc >> 6, dh = gc & (DHh - 1);
                    out0[((size_t)h * RPAD + gr) * DHh + dh] = (bf16_t)v;
                }
            }
        }
    }
}

// ---------------------------------------------------------------------------
// Fused relative attention, flash-style online softmax.
// Grid: B*H*(Q/128) blocks; 8 waves; wave = 16 q-rows; K tiles of 32.
// score = (q_ac.k + q_bd.r_head_k[j+1]) * SCALE, causal mask, softmax, P.V^T
// ---------------------------------------------------------------------------
__global__ __launch_bounds__(256) void attn_k(
    const bf16_t* __restrict__ qac, const bf16_t* __restrict__ qbd,
    const bf16_t* __restrict__ kh,  const bf16_t* __restrict__ vt,
    const bf16_t* __restrict__ rhk, bf16_t* __restrict__ avec) {

    __shared__ bf16_t plds[8][16][32];     // per-wave P bounce buffer (8 KB)

    const int lane = threadIdx.x & 31;
    const int wid  = threadIdx.x >> 5;
    const int lrow = lane & 15, lh = lane >> 4;

    const int blk = blockIdx.x;
    const int qb  = blk & 7;               // q tile of 128
    const int bh  = blk >> 3;              // b*H + h
    const int h   = bh & (Hh - 1);
    const int b   = bh >> 4;
    const int q0  = qb * 128 + wid * 16;

    // Per-wave A fragments (fixed q rows), two K-steps over Dh=64
    const bf16_t* qacp = qac + ((size_t)bh * Qq + q0 + lrow) * DHh;
    const bf16_t* qbdp = qbd + ((size_t)bh * Qq + q0 + lrow) * DHh;
    v16bf aac[2], abd[2];
#pragma unroll
    for (int ks = 0; ks < 2; ++ks) {
        aac[ks] = load_a_frag(qacp + ks * 32, lh);
        abd[ks] = load_a_frag(qbdp + ks * 32, lh);
    }

    const bf16_t* kbase = kh  + (size_t)bh * Qq * DHh;         // [k][dh]
    const bf16_t* rbase = rhk + (size_t)h * RPAD * DHh;        // [k][dh], +1 pad row
    const bf16_t* vbase = vt  + (size_t)bh * DHh * Qq;         // [dh][k]

    float mstat[8], lstat[8];
#pragma unroll
    for (int i = 0; i < 8; ++i) { mstat[i] = NEG_INF; lstat[i] = 0.f; }
    v8f o[4];
#pragma unroll
    for (int nt = 0; nt < 4; ++nt) o[nt] = v8zero();

    const int jend = qb * 128 + 128;       // block-uniform causal bound

    for (int j0 = 0; j0 < jend; j0 += 32) {
        // ---- scores: AC + BD(shifted) for a 16x32 strip -------------------
        v8f s[2];
#pragma unroll
        for (int nt = 0; nt < 2; ++nt) {
            s[nt] = v8zero();
            const bf16_t* krow = kbase + (size_t)(j0 + nt * 16 + lrow) * DHh + 16 * lh;
            s[nt] = wmma_bf16(aac[0], load_b_frag(krow),      s[nt]);
            s[nt] = wmma_bf16(aac[1], load_b_frag(krow + 32), s[nt]);
            const bf16_t* rrow = rbase + (size_t)(j0 + 1 + nt * 16 + lrow) * DHh + 16 * lh;
            s[nt] = wmma_bf16(abd[0], load_b_frag(rrow),      s[nt]);
            s[nt] = wmma_bf16(abd[1], load_b_frag(rrow + 32), s[nt]);
        }

        // ---- scale + causal mask + row max --------------------------------
        float cmax[8];
#pragma unroll
        for (int i = 0; i < 8; ++i) {
            const int row = q0 + 8 * lh + i;
            float v0 = s[0][i] * SCALE;
            float v1 = s[1][i] * SCALE;
            if (j0 + lrow      > row) v0 = NEG_INF;
            if (j0 + 16 + lrow > row) v1 = NEG_INF;
            s[0][i] = v0; s[1][i] = v1;
            cmax[i] = fmaxf(v0, v1);
        }
#pragma unroll
        for (int i = 0; i < 8; ++i)
#pragma unroll
            for (int d = 1; d < 16; d <<= 1)
                cmax[i] = fmaxf(cmax[i], __shfl_xor(cmax[i], d));

        // ---- online softmax update ----------------------------------------
        float fac[8], rsum[8];
#pragma unroll
        for (int i = 0; i < 8; ++i) {
            const float mn = fmaxf(mstat[i], cmax[i]);
            fac[i] = __expf(mstat[i] - mn);
            mstat[i] = mn;
            const float p0 = __expf(s[0][i] - mn);
            const float p1 = __expf(s[1][i] - mn);
            s[0][i] = p0; s[1][i] = p1;
            rsum[i] = p0 + p1;
        }
#pragma unroll
        for (int i = 0; i < 8; ++i)
#pragma unroll
            for (int d = 1; d < 16; d <<= 1)
                rsum[i] += __shfl_xor(rsum[i], d);
#pragma unroll
        for (int i = 0; i < 8; ++i) lstat[i] = lstat[i] * fac[i] + rsum[i];
#pragma unroll
        for (int nt = 0; nt < 4; ++nt)
#pragma unroll
            for (int i = 0; i < 8; ++i) o[nt][i] *= fac[i];

        // ---- P: D-layout -> A-layout via per-wave LDS bounce --------------
#pragma unroll
        for (int nt = 0; nt < 2; ++nt)
#pragma unroll
            for (int i = 0; i < 8; ++i)
                plds[wid][8 * lh + i][nt * 16 + lrow] = (bf16_t)s[nt][i];
        __syncthreads();
        const v16bf pa = load_a_frag(&plds[wid][lrow][0], lh);
        __syncthreads();

        // ---- O += P @ V  (V stored transposed: [dh][k], contiguous K) -----
#pragma unroll
        for (int nt = 0; nt < 4; ++nt) {
            const bf16_t* vrow = vbase + (size_t)(nt * 16 + lrow) * Qq + j0 + 16 * lh;
            o[nt] = wmma_bf16(pa, load_b_frag(vrow), o[nt]);
        }
    }

    // ---- epilogue: normalize, write attn_vec bf16 [B*Q, H*Dh] -------------
    float inv[8];
#pragma unroll
    for (int i = 0; i < 8; ++i) inv[i] = 1.0f / lstat[i];
#pragma unroll
    for (int nt = 0; nt < 4; ++nt)
#pragma unroll
        for (int i = 0; i < 8; ++i) {
            const int row = q0 + 8 * lh + i;
            const int col = h * DHh + nt * 16 + lrow;
            avec[((size_t)b * Qq + row) * Dd + col] = (bf16_t)(o[nt][i] * inv[i]);
        }
}

// ---------------------------------------------------------------------------
// Residual + LayerNorm: out = LN(w + attn_out) * gamma + beta
// One block (256 thr) per token row of 1024.
// ---------------------------------------------------------------------------
__global__ __launch_bounds__(256) void ln_k(
    const float* __restrict__ w, const float* __restrict__ aout,
    const float* __restrict__ gamma, const float* __restrict__ beta,
    float* __restrict__ out) {

    __shared__ float redA[8], redB[8];
    const int row = blockIdx.x;
    const int t = threadIdx.x;
    const float* pw = w    + (size_t)row * Dd;
    const float* pa = aout + (size_t)row * Dd;

    float x[4], s = 0.f, s2 = 0.f;
#pragma unroll
    for (int k = 0; k < 4; ++k) {
        const int c = t + k * 256;
        x[k] = pw[c] + pa[c];
        s += x[k]; s2 += x[k] * x[k];
    }
#pragma unroll
    for (int d = 1; d < 32; d <<= 1) { s += __shfl_xor(s, d); s2 += __shfl_xor(s2, d); }
    if ((t & 31) == 0) { redA[t >> 5] = s; redB[t >> 5] = s2; }
    __syncthreads();
    float ts = 0.f, ts2 = 0.f;
#pragma unroll
    for (int i = 0; i < 8; ++i) { ts += redA[i]; ts2 += redB[i]; }
    const float mu = ts * (1.0f / Dd);
    const float var = ts2 * (1.0f / Dd) - mu * mu;
    const float rs = rsqrtf(var + 1e-5f);
#pragma unroll
    for (int k = 0; k < 4; ++k) {
        const int c = t + k * 256;
        out[(size_t)row * Dd + c] = (x[k] - mu) * rs * gamma[c] + beta[c];
    }
}

// ---------------------------------------------------------------------------
// Host launcher
// ---------------------------------------------------------------------------
extern "C" void kernel_launch(void* const* d_in, const int* in_sizes, int n_in,
                              void* d_out, int out_size, void* d_ws, size_t ws_size,
                              hipStream_t stream) {
    (void)in_sizes; (void)n_in; (void)out_size; (void)ws_size;

    const float* w   = (const float*)d_in[0];
    const float* r   = (const float*)d_in[1];
    const float* rwb = (const float*)d_in[2];
    const float* rrb = (const float*)d_in[3];
    // d_in[4] attn_mask: pure causal tril -> applied analytically in attn_k
    const float* Wq  = (const float*)d_in[5];
    const float* Wk  = (const float*)d_in[6];
    const float* Wv  = (const float*)d_in[7];
    const float* Wr  = (const float*)d_in[8];
    const float* Wo  = (const float*)d_in[9];
    const float* gam = (const float*)d_in[10];
    const float* bet = (const float*)d_in[11];
    float* out = (float*)d_out;

    char* ws = (char*)d_ws;
    size_t off = 0;
    auto take = [&](size_t bytes) -> char* {
        char* p = ws + off;
        off += (bytes + 255) & ~(size_t)255;
        return p;
    };
    const size_t R = (size_t)Bb * Qq;                       // 4096
    bf16_t* w_bf  = (bf16_t*)take(R * Dd * 2);
    bf16_t* r_bf  = (bf16_t*)take((size_t)Qq * Dd * 2);
    bf16_t* wq_bf = (bf16_t*)take((size_t)Dd * Dd * 2);
    bf16_t* wk_bf = (bf16_t*)take((size_t)Dd * Dd * 2);
    bf16_t* wv_bf = (bf16_t*)take((size_t)Dd * Dd * 2);
    bf16_t* wr_bf = (bf16_t*)take((size_t)Dd * Dd * 2);
    bf16_t* wo_bf = (bf16_t*)take((size_t)Dd * Dd * 2);
    bf16_t* qac   = (bf16_t*)take(R * Dd * 2);
    bf16_t* qbd   = (bf16_t*)take(R * Dd * 2);
    bf16_t* khd   = (bf16_t*)take(R * Dd * 2);
    bf16_t* vtb   = (bf16_t*)take(R * Dd * 2);
    bf16_t* rhk   = (bf16_t*)take((size_t)Hh * RPAD * DHh * 2);
    bf16_t* avec  = (bf16_t*)take(R * Dd * 2);
    float*  aout  = (float*)take(R * Dd * 4);

    auto cvt = [&](const float* src, bf16_t* dst, int n) {
        const int n4 = n / 4;
        cvt_k<<<(n4 + 255) / 256, 256, 0, stream>>>(src, dst, n4);
    };
    cvt(w,  w_bf,  (int)(R * Dd));
    cvt(r,  r_bf,  Qq * Dd);
    cvt(Wq, wq_bf, Dd * Dd);
    cvt(Wk, wk_bf, Dd * Dd);
    cvt(Wv, wv_bf, Dd * Dd);
    cvt(Wr, wr_bf, Dd * Dd);
    cvt(Wo, wo_bf, Dd * Dd);

    // Projections (WMMA bf16 GEMMs with fused head-layout epilogues)
    gemm_k<1><<<dim3((int)(R / 128), 8), 256, 0, stream>>>(w_bf, wq_bf, nullptr, qac, qbd, rwb, rrb);
    gemm_k<2><<<dim3((int)(R / 128), 8), 256, 0, stream>>>(w_bf, wk_bf, nullptr, khd, nullptr, nullptr, nullptr);
    gemm_k<3><<<dim3((int)(R / 128), 8), 256, 0, stream>>>(w_bf, wv_bf, nullptr, vtb, nullptr, nullptr, nullptr);
    gemm_k<4><<<dim3(Qq / 128, 8), 256, 0, stream>>>(r_bf, wr_bf, nullptr, rhk, nullptr, nullptr, nullptr);
    zero_pad_k<<<64, 256, 0, stream>>>(rhk);

    // Fused relative attention
    attn_k<<<Bb * Hh * (Qq / 128), 256, 0, stream>>>(qac, qbd, khd, vtb, rhk, avec);

    // Output projection + residual LayerNorm
    gemm_k<0><<<dim3((int)(R / 128), 8), 256, 0, stream>>>(avec, wo_bf, aout, nullptr, nullptr, nullptr, nullptr);
    ln_k<<<(int)R, 256, 0, stream>>>(w, aout, gam, bet, out);
}
